// Model_8443905704512
// MI455X (gfx1250) — compile-verified, matrix operand order checked by reading, not run on hardware
//
#include <hip/hip_runtime.h>

#define HDIM 512
#define BSZ 128
#define NGATE 2048            // 4*HDIM
#define TSTEPS 201
#define NCHAR 100

typedef __attribute__((ext_vector_type(16))) _Float16 v16h;
typedef __attribute__((ext_vector_type(8)))  float    v8f;
typedef __attribute__((ext_vector_type(4)))  unsigned int v4u;

union FragH { v16h v; v4u u[2]; };

// Load one 16x32 f16 WMMA A/B fragment. p must already point at
// row*ld + k0 + ((lane>>4)<<3). ISA layout (CDNA5 ISA): per lane the 16
// halves are K = {kb..kb+7, kb+16..kb+23}.
__device__ __forceinline__ v16h ldfrag(const _Float16* __restrict__ p) {
  FragH f;
  f.u[0] = *(const v4u*)(p);        // halves 0..7
  f.u[1] = *(const v4u*)(p + 16);   // halves 8..15
  return f.v;
}

__device__ __forceinline__ v8f wmma_f16(v16h a, v16h b, v8f c) {
  return __builtin_amdgcn_wmma_f32_16x16x32_f16(false, a, false, b,
                                                (short)0, c, false, false);
}

__device__ __forceinline__ float sigm(float x) { return 1.0f / (1.0f + expf(-x)); }

// ---- fused gates-GEMM + LSTM cell, both directions of one layer ----
// Wave computes a 32(M) x 16(j) tile of ALL FOUR gates: B column-tiles at
// rows {j, j+512, j+1024, j+1536} of Wcat, so i/f/g/o stay in-register and
// the activation runs in the epilogue (no gates buffer, no pointwise pass).
// X: [B,KX] f16; Hold: [2][B,H] f16 (read); Hnew: [2][B,H] f16 (write, ping-pong);
// W: [2][2048, KX+512] f16; bsum: [2][2048] f32; Cf: [2][B,H] f32 (in-place).
// grid = (HDIM/128, B/32, 2), block = 256 (8 waves, one j-tile each).
template <int KX>
__global__ void __launch_bounds__(256)
gemm_lstm_fused(const _Float16* __restrict__ X,
                const _Float16* __restrict__ Hold,
                const _Float16* __restrict__ W,
                const float* __restrict__ bsum,
                float* __restrict__ Cf,
                _Float16* __restrict__ Hnew,
                _Float16* __restrict__ X16,
                float* __restrict__ X32) {
  constexpr int KTOT = KX + HDIM;
  const int dir  = blockIdx.z;
  const int lane = threadIdx.x & 31;
  const int wave = threadIdx.x >> 5;
  const int j16  = (blockIdx.x * 8 + wave) * 16;   // j column tile
  const int mb   = blockIdx.y * 32;                // batch-row block

  const _Float16* Hd = Hold + (size_t)dir * BSZ * HDIM;
  const _Float16* Wd = W    + (size_t)dir * NGATE * KTOT;
  const float*    bs = bsum + (size_t)dir * NGATE;
  float*          Cd = Cf   + (size_t)dir * BSZ * HDIM;
  _Float16*       Hn = Hnew + (size_t)dir * BSZ * HDIM;

  const int koff = (lane >> 4) << 3;               // per-lane K sub-offset
  const int ar0  = mb + (lane & 15);

  // A row pointers (two M halves), B row pointers (four gates).
  const _Float16* pa0 = X + (size_t)ar0 * KX + koff;
  const _Float16* pa1 = X + (size_t)(ar0 + 16) * KX + koff;
  const _Float16* pb[4];
#pragma unroll
  for (int g = 0; g < 4; ++g)
    pb[g] = Wd + (size_t)(g * HDIM + j16 + (lane & 15)) * KTOT + koff;

  v8f acc[2][4] = {};

  // Phase 1: x @ W_ih^T   (constant trip count -> pure pointer increments)
  for (int k0 = 0; k0 < KX; k0 += 32) {
    v16h a0 = ldfrag(pa0 + k0);
    v16h a1 = ldfrag(pa1 + k0);
    __builtin_prefetch(pb[0] + k0 + 256, 0, 1);
#pragma unroll
    for (int g = 0; g < 4; ++g) {
      v16h b = ldfrag(pb[g] + k0);
      acc[0][g] = wmma_f16(a0, b, acc[0][g]);
      acc[1][g] = wmma_f16(a1, b, acc[1][g]);
    }
  }
  // Phase 2: h @ W_hh^T
  const _Float16* ph0 = Hd + (size_t)ar0 * HDIM + koff;
  const _Float16* ph1 = Hd + (size_t)(ar0 + 16) * HDIM + koff;
  for (int k0 = 0; k0 < HDIM; k0 += 32) {
    v16h a0 = ldfrag(ph0 + k0);
    v16h a1 = ldfrag(ph1 + k0);
#pragma unroll
    for (int g = 0; g < 4; ++g) {
      v16h b = ldfrag(pb[g] + KX + k0);
      acc[0][g] = wmma_f16(a0, b, acc[0][g]);
      acc[1][g] = wmma_f16(a1, b, acc[1][g]);
    }
  }

  // Epilogue: bias + LSTM activation, in-register per element.
  // C/D layout: element (VGPR r, lane) = (M = r + 8*(lane>=16), N = lane&15).
  const int jc = j16 + (lane & 15);
  const float bi = bs[jc];
  const float bf = bs[HDIM + jc];
  const float bg = bs[2 * HDIM + jc];
  const float bo = bs[3 * HDIM + jc];
#pragma unroll
  for (int mh = 0; mh < 2; ++mh) {
    const int m0 = mb + mh * 16 + ((lane >> 4) << 3);
#pragma unroll
    for (int r = 0; r < 8; ++r) {
      const int m = m0 + r;
      const float iv = acc[mh][0][r] + bi;
      const float fv = acc[mh][1][r] + bf;
      const float gv = acc[mh][2][r] + bg;
      const float ov = acc[mh][3][r] + bo;
      const size_t si = (size_t)m * HDIM + jc;
      const float c2 = sigm(fv) * Cd[si] + sigm(iv) * tanhf(gv);
      const float h2 = sigm(ov) * tanhf(c2);
      Cd[si] = c2;
      Hn[si] = (_Float16)h2;
      const size_t xo = (size_t)m * (2 * HDIM) + (size_t)dir * HDIM + jc;
      if (X16) X16[xo] = (_Float16)h2;
      if (X32) X32[xo] = h2;
    }
  }
}

// -------- LayerNorm + projection + argmax + embedding feedback --------
__global__ void __launch_bounds__(128)
ln_proj_argmax(const float* __restrict__ Xc, const float* __restrict__ lng,
               const float* __restrict__ lnb, const float* __restrict__ pw,
               const float* __restrict__ pb, const _Float16* __restrict__ emb,
               _Float16* __restrict__ X0, float* __restrict__ out, int t) {
  __shared__ float sy[2 * HDIM];
  __shared__ float red[128];
  __shared__ int   ridx[128];
  const int b = blockIdx.x, tid = threadIdx.x;
  const float* xr = Xc + (size_t)b * (2 * HDIM);

  float s = 0.0f;
  for (int i = tid; i < 2 * HDIM; i += 128) { float v = xr[i]; sy[i] = v; s += v; }
  red[tid] = s;
  __syncthreads();
  for (int o = 64; o > 0; o >>= 1) { if (tid < o) red[tid] += red[tid + o]; __syncthreads(); }
  const float mu = red[0] * (1.0f / (2 * HDIM));
  __syncthreads();

  float vs = 0.0f;
  for (int i = tid; i < 2 * HDIM; i += 128) { float dv = sy[i] - mu; vs += dv * dv; }
  red[tid] = vs;
  __syncthreads();
  for (int o = 64; o > 0; o >>= 1) { if (tid < o) red[tid] += red[tid + o]; __syncthreads(); }
  const float rstd = rsqrtf(red[0] * (1.0f / (2 * HDIM)) + 1e-5f);
  __syncthreads();

  for (int i = tid; i < 2 * HDIM; i += 128)
    sy[i] = (sy[i] - mu) * rstd * lng[i] + lnb[i];
  __syncthreads();

  float lg = -__builtin_inff();
  if (tid < NCHAR) {
    float a = pb[tid];
    const float* w = pw + (size_t)tid * (2 * HDIM);
    for (int k = 0; k < 2 * HDIM; ++k) a += sy[k] * w[k];
    lg = a;
    out[(size_t)b * NCHAR * TSTEPS + (size_t)tid * TSTEPS + t] = a;
  }
  red[tid] = lg; ridx[tid] = tid;
  __syncthreads();
  for (int o = 64; o > 0; o >>= 1) {
    if (tid < o) {
      if (red[tid + o] > red[tid] ||
          (red[tid + o] == red[tid] && ridx[tid + o] < ridx[tid])) {
        red[tid] = red[tid + o]; ridx[tid] = ridx[tid + o];
      }
    }
    __syncthreads();
  }
  const int best = ridx[0];
  const _Float16* er = emb + (size_t)best * HDIM;
  _Float16* xo = X0 + (size_t)b * HDIM;
  for (int i = tid; i < HDIM; i += 128) xo[i] = er[i];
}

// -------- prep kernels --------
__global__ void convert_pack(const float* __restrict__ src, _Float16* __restrict__ dst,
                             int kc, int dstld, int dstcol, int total) {
  const int tid = blockIdx.x * blockDim.x + threadIdx.x;
  if (tid >= total) return;
  const int r = tid / kc, c = tid - r * kc;
  dst[(size_t)r * dstld + dstcol + c] = (_Float16)src[tid];
}

__global__ void bias_sum_k(const float* __restrict__ a, const float* __restrict__ b,
                           float* __restrict__ d, int n) {
  const int tid = blockIdx.x * blockDim.x + threadIdx.x;
  if (tid < n) d[tid] = a[tid] + b[tid];
}

__global__ void conv_f16_k(const float* __restrict__ s, _Float16* __restrict__ d, int n) {
  const int tid = blockIdx.x * blockDim.x + threadIdx.x;
  if (tid < n) d[tid] = (_Float16)s[tid];
}

__global__ void init_state(const float* __restrict__ h0, const float* __restrict__ c0,
                           _Float16* __restrict__ Hh, float* __restrict__ Cf) {
  const int tid = blockIdx.x * blockDim.x + threadIdx.x;  // < 6*B*H
  const int id = tid >> 16;
  const int j  = tid & (HDIM - 1);
  Hh[tid] = (_Float16)h0[id * HDIM + j];
  Cf[tid] = c0[id * HDIM + j];
}

__global__ void init_x0(const _Float16* __restrict__ emb, _Float16* __restrict__ X0) {
  const int tid = blockIdx.x * blockDim.x + threadIdx.x;  // < B*H
  X0[tid] = emb[tid & (HDIM - 1)];                        // embed row 0 (<SOS>)
}

extern "C" void kernel_launch(void* const* d_in, const int* in_sizes, int n_in,
                              void* d_out, int out_size, void* d_ws, size_t ws_size,
                              hipStream_t stream) {
  (void)in_sizes; (void)n_in; (void)out_size; (void)ws_size;
  const float* w_ih[3] = {(const float*)d_in[1], (const float*)d_in[3], (const float*)d_in[5]};
  const float* w_hh[3] = {(const float*)d_in[2], (const float*)d_in[4], (const float*)d_in[6]};
  const float* b_ih   = (const float*)d_in[7];
  const float* b_hh   = (const float*)d_in[8];
  const float* ln_g   = (const float*)d_in[9];
  const float* ln_b   = (const float*)d_in[10];
  const float* proj_w = (const float*)d_in[11];
  const float* proj_b = (const float*)d_in[12];
  const float* embed  = (const float*)d_in[13];
  const float* h0     = (const float*)d_in[14];
  const float* c0     = (const float*)d_in[15];
  float* out = (float*)d_out;

  // workspace layout (all slices 256B aligned)
  char* ws = (char*)d_ws;
  size_t off = 0;
  auto take = [&](size_t bytes) -> char* {
    char* p = ws + off;
    off = (off + bytes + 255) & ~(size_t)255;
    return p;
  };
  const int Kin[3] = {HDIM, 2 * HDIM, 2 * HDIM};
  _Float16* Wcat[3];
  for (int l = 0; l < 3; ++l)
    Wcat[l] = (_Float16*)take((size_t)2 * NGATE * (Kin[l] + HDIM) * sizeof(_Float16));
  float*    bsum  = (float*)take((size_t)3 * 2 * NGATE * sizeof(float));
  _Float16* emb16 = (_Float16*)take((size_t)NCHAR * HDIM * sizeof(_Float16));
  _Float16* X0    = (_Float16*)take((size_t)BSZ * HDIM * sizeof(_Float16));
  _Float16* X1    = (_Float16*)take((size_t)BSZ * 2 * HDIM * sizeof(_Float16));
  _Float16* X2    = (_Float16*)take((size_t)BSZ * 2 * HDIM * sizeof(_Float16));
  float*    Xc    = (float*)take((size_t)BSZ * 2 * HDIM * sizeof(float));
  _Float16* HhA   = (_Float16*)take((size_t)6 * BSZ * HDIM * sizeof(_Float16));
  _Float16* HhB   = (_Float16*)take((size_t)6 * BSZ * HDIM * sizeof(_Float16));
  float*    Cf    = (float*)take((size_t)6 * BSZ * HDIM * sizeof(float));

  // ---- prep: pack weights to f16 [2048, Kx+H] per (layer,dir); fold biases ----
  for (int l = 0; l < 3; ++l) {
    const int Ktot = Kin[l] + HDIM;
    for (int d = 0; d < 2; ++d) {
      const float* sih = w_ih[l] + (size_t)d * NGATE * Kin[l];
      const float* shh = w_hh[l] + (size_t)d * NGATE * HDIM;
      _Float16* dst = Wcat[l] + (size_t)d * NGATE * Ktot;
      const int tih = NGATE * Kin[l];
      convert_pack<<<(tih + 255) / 256, 256, 0, stream>>>(sih, dst, Kin[l], Ktot, 0, tih);
      const int thh = NGATE * HDIM;
      convert_pack<<<(thh + 255) / 256, 256, 0, stream>>>(shh, dst, HDIM, Ktot, Kin[l], thh);
    }
  }
  bias_sum_k<<<(3 * 2 * NGATE + 255) / 256, 256, 0, stream>>>(b_ih, b_hh, bsum, 3 * 2 * NGATE);
  conv_f16_k<<<(NCHAR * HDIM + 255) / 256, 256, 0, stream>>>(embed, emb16, NCHAR * HDIM);
  init_state<<<(6 * BSZ * HDIM) / 256, 256, 0, stream>>>(h0, c0, HhA, Cf);
  init_x0<<<(BSZ * HDIM) / 256, 256, 0, stream>>>(emb16, X0);

  // ---- sequential decode: 4 launches per step ----
  const dim3 fgrid(HDIM / 128, BSZ / 32, 2);   // (4,4,2), 256 threads
  _Float16* Xin[3]  = {X0, X1, X2};
  _Float16* Xo16[3] = {X1, X2, nullptr};
  _Float16* Hcur = HhA;
  _Float16* Hnxt = HhB;
  for (int t = 0; t < TSTEPS; ++t) {
    for (int l = 0; l < 3; ++l) {
      const size_t so = (size_t)(2 * l) * BSZ * HDIM;
      if (l == 0) {
        gemm_lstm_fused<HDIM><<<fgrid, 256, 0, stream>>>(
            Xin[l], Hcur + so, Wcat[l], bsum + (size_t)l * 2 * NGATE,
            Cf + so, Hnxt + so, Xo16[l], nullptr);
      } else {
        gemm_lstm_fused<2 * HDIM><<<fgrid, 256, 0, stream>>>(
            Xin[l], Hcur + so, Wcat[l], bsum + (size_t)l * 2 * NGATE,
            Cf + so, Hnxt + so, Xo16[l], (l == 2) ? Xc : nullptr);
      }
    }
    ln_proj_argmax<<<BSZ, 128, 0, stream>>>(Xc, ln_g, ln_b, proj_w, proj_b,
                                            emb16, X0, out, t);
    _Float16* tmp = Hcur; Hcur = Hnxt; Hnxt = tmp;
  }
}
